// ContrastiveDGINConv_23330262352383
// MI455X (gfx1250) — matrix operand order repaired and checked by reading, not compiled, not run on hardware
//
#include <hip/hip_runtime.h>
#include <hip/hip_bf16.h>
#include <stdint.h>
#include <stddef.h>

typedef __bf16 bf16_t;
typedef __attribute__((ext_vector_type(16))) __bf16 v16bf;
typedef __attribute__((ext_vector_type(8)))  float  v8f;
typedef __attribute__((ext_vector_type(4)))  int    v4i;

#define Uf 128
#define Nn 2048
#define Ne 32768
#define Nb 8
#define Nd 3

union BF16x16 { uint4 q[2]; v16bf v; };

__device__ __forceinline__ v8f wmma_bf16(v16bf a, v16bf b, v8f c) {
  return __builtin_amdgcn_wmma_f32_16x16x32_bf16(false, a, false, b, (short)0, c, false, false);
}

// ---- async global->LDS copy (gfx1250 GLOBAL_LOAD_ASYNC_TO_LDS_B128), guarded ----
#if __has_builtin(__builtin_amdgcn_global_load_async_to_lds_b128)
#define HAVE_ASYNC_LDS 1
typedef __attribute__((address_space(1))) v4i* g_v4i_ptr;
typedef __attribute__((address_space(3))) v4i* l_v4i_ptr;
__device__ __forceinline__ void copy16_g2l(bf16_t* lds_dst, const bf16_t* gsrc) {
  void* g = (void*)gsrc;   // drop const
  void* l = (void*)lds_dst;
  __builtin_amdgcn_global_load_async_to_lds_b128((g_v4i_ptr)g, (l_v4i_ptr)l, 0, 0);
}
__device__ __forceinline__ void async_wait() {
#if __has_builtin(__builtin_amdgcn_s_wait_asynccnt)
  __builtin_amdgcn_s_wait_asynccnt(0);
#else
  asm volatile("s_wait_asynccnt 0" ::: "memory");
#endif
}
#else
#define HAVE_ASYNC_LDS 0
__device__ __forceinline__ void copy16_g2l(bf16_t* lds_dst, const bf16_t* gsrc) {
  *(uint4*)lds_dst = *(const uint4*)gsrc;
}
__device__ __forceinline__ void async_wait() {}
#endif

// One wave computes a 16x32 output tile (two 16x16 WMMA tiles) of A[16,KTOT] @ B[KTOT,128].
// A: LDS, row-major bf16, row stride KTOT (16B aligned rows).
// BT: global, packed column-major bf16: BT[n*KTOT + k].
template <int KTOT>
__device__ __forceinline__ void gemm2(const bf16_t* A, const bf16_t* __restrict__ BT,
                                      int lane, int n0, v8f& c0, v8f& c1) {
  const int hi = lane >> 4, lo = lane & 15;
  #pragma unroll
  for (int kt = 0; kt < KTOT / 32; ++kt) {
    const int k0 = kt * 32;
    BF16x16 Af, B0, B1;
    const bf16_t* ap = A + lo * KTOT + k0 + hi * 8;
    Af.q[0] = *(const uint4*)ap;
    Af.q[1] = *(const uint4*)(ap + 16);
    const bf16_t* bp = BT + (size_t)(n0 + lo) * KTOT + k0 + hi * 8;
    B0.q[0] = *(const uint4*)bp;  B0.q[1] = *(const uint4*)(bp + 16);
    const bf16_t* bp1 = bp + (size_t)16 * KTOT;
    B1.q[0] = *(const uint4*)bp1; B1.q[1] = *(const uint4*)(bp1 + 16);
    c0 = wmma_bf16(Af.v, B0.v, c0);
    c1 = wmma_bf16(Af.v, B1.v, c1);
  }
}

// ---------------- prep kernels ----------------
__global__ void cvt_bf16_kernel(const float* __restrict__ in, bf16_t* __restrict__ out, int n) {
  int i = blockIdx.x * 256 + threadIdx.x;
  if (i < n) out[i] = (bf16_t)in[i];
}

__global__ void pack_wT_kernel(const float* __restrict__ W, bf16_t* __restrict__ WT, int K, int Nc) {
  int i = blockIdx.x * 256 + threadIdx.x;
  if (i >= K * Nc) return;
  int k = i / Nc, n = i % Nc;
  WT[(size_t)n * K + k] = (bf16_t)W[i];
}

// norm_params [D][4][4][U] -> st [D][4][2][U] : s = g*rsqrt(var+eps), t = beta - mean*s
__global__ void norm_prep_kernel(const float* __restrict__ np, float* __restrict__ st) {
  int i = blockIdx.x * 256 + threadIdx.x;
  if (i >= Nd * 4 * Uf) return;
  int l = i / (4 * Uf); int r = i % (4 * Uf); int j = r / Uf; int c = r % Uf;
  const float* p = np + ((size_t)(l * 4 + j) * 4) * Uf;
  float gamma = p[c], beta = p[Uf + c], mean = p[2 * Uf + c], var = p[3 * Uf + c];
  float s = gamma * rsqrtf(var + 1e-3f);
  float* o = st + ((size_t)(l * 4 + j) * 2) * Uf;
  o[c] = s; o[Uf + c] = beta - mean * s;
}

__global__ void zero_kernel(float4* __restrict__ p, int n4) {
  int i = blockIdx.x * 256 + threadIdx.x;
  if (i < n4) p[i] = make_float4(0.f, 0.f, 0.f, 0.f);
}

// ---------------- edge MLP + fused scatter ----------------
// grid: Nb * (Ne/64), block 128 (4 waves). Each WG: 64 edges x 128 out channels.
// Each wave: 4 M-subtiles x 2 N-tiles; B fragments loaded once per k-step, reused 4x.
__global__ __launch_bounds__(128) void edge_mlp_kernel(
    const bf16_t* __restrict__ nf, const bf16_t* __restrict__ ef,
    const int* __restrict__ eidx, const int* __restrict__ erev,
    const bf16_t* __restrict__ WeT, const float* __restrict__ bias,
    const float* __restrict__ st,
    bf16_t* __restrict__ eu, float* __restrict__ agg) {
  __shared__ __align__(16) bf16_t ecA[64 * 384];
  __shared__ int ssrc[64], sdst[64], srev[64];
  const int tid = threadIdx.x, lane = tid & 31, wave = tid >> 5;
  const int b = blockIdx.x >> 9, et = blockIdx.x & 511, e0 = et * 64;
  if (tid < 64) {
    const int e = b * Ne + e0 + tid;
    srev[tid] = erev[e];
    ssrc[tid] = eidx[e * 2 + 1];   // GatherNodesOutgoing -> col 1
    sdst[tid] = eidx[e * 2 + 0];   // GatherNodesIngoing  -> col 0
  }
  __syncthreads();
  // Stage ec = [src | dst | ef] as 64x384 bf16, 16B chunks (async to LDS when available).
  for (int c = tid; c < 64 * 48; c += 128) {
    int r = c / 48, seg = c % 48, col8 = (seg & 15) * 8;
    const bf16_t* sp; int doff;
    if (seg < 16)      { sp = nf + ((size_t)(b * Nn + ssrc[r])) * Uf + col8; doff = col8; }
    else if (seg < 32) { sp = nf + ((size_t)(b * Nn + sdst[r])) * Uf + col8; doff = 128 + col8; }
    else               { sp = ef + ((size_t)(b * Ne + e0 + r)) * Uf + col8; doff = 256 + col8; }
    copy16_g2l(&ecA[r * 384 + doff], sp);
  }
  async_wait();
  __syncthreads();

  const int n0 = wave * 32, hi = lane >> 4, lo = lane & 15;
  v8f acc[4][2];
  #pragma unroll
  for (int ms = 0; ms < 4; ++ms) { acc[ms][0] = {}; acc[ms][1] = {}; }
  const bf16_t* bbase = WeT + (size_t)(n0 + lo) * 384 + hi * 8;
  #pragma unroll
  for (int kt = 0; kt < 12; ++kt) {
    const int k0 = kt * 32;
    BF16x16 B0, B1;
    const bf16_t* bp = bbase + k0;
    B0.q[0] = *(const uint4*)bp;  B0.q[1] = *(const uint4*)(bp + 16);
    const bf16_t* bp1 = bp + 16 * 384;
    B1.q[0] = *(const uint4*)bp1; B1.q[1] = *(const uint4*)(bp1 + 16);
    #pragma unroll
    for (int ms = 0; ms < 4; ++ms) {
      BF16x16 Af;
      const bf16_t* ap = &ecA[(ms * 16 + lo) * 384 + k0 + hi * 8];
      Af.q[0] = *(const uint4*)ap;
      Af.q[1] = *(const uint4*)(ap + 16);
      acc[ms][0] = wmma_bf16(Af.v, B0.v, acc[ms][0]);
      acc[ms][1] = wmma_bf16(Af.v, B1.v, acc[ms][1]);
    }
  }
  #pragma unroll
  for (int t = 0; t < 2; ++t) {
    const int n = n0 + t * 16 + lo;
    const float bn = bias[n], s = st[n], tt = st[Uf + n];
    #pragma unroll
    for (int ms = 0; ms < 4; ++ms) {
      #pragma unroll
      for (int r8 = 0; r8 < 8; ++r8) {
        const int m = ms * 16 + r8 + hi * 8;
        float y = fmaxf(acc[ms][t][r8] + bn, 0.f);   // relu(x@W + b)
        y = fmaxf(s * y + tt, 0.f);                  // relu(norm(.))
        eu[((size_t)(b * Ne + e0 + m)) * Uf + n] = (bf16_t)y;
        atomicAdd(&agg[((size_t)(b * Nn + srev[m])) * Uf + n], y);
      }
    }
  }
}

// ---------------- fused node MLP chain ----------------
// grid: Nb * (Nn/16), block 128 (4 waves). Three chained GEMMs through LDS.
__global__ __launch_bounds__(128) void node_mlp_kernel(
    const bf16_t* __restrict__ nf, const float* __restrict__ agg,
    const bf16_t* __restrict__ Wg1T, const float* __restrict__ bg1, const float* __restrict__ st1,
    const bf16_t* __restrict__ Wg2T, const float* __restrict__ bg2, const float* __restrict__ st2,
    const float* __restrict__ st3,
    const bf16_t* __restrict__ WuT,  const float* __restrict__ bu,
    bf16_t* __restrict__ nf_out) {
  __shared__ __align__(16) bf16_t A0[16 * 256];
  __shared__ __align__(16) bf16_t A1[16 * 128];
  const int tid = threadIdx.x, lane = tid & 31, wave = tid >> 5;
  const int b = blockIdx.x >> 7, nt = blockIdx.x & 127, m0 = nt * 16;
  // Stage nc = [nf | agg->bf16] as 16x256 bf16.
  for (int c = tid; c < 16 * 32; c += 128) {
    int r = c >> 5, seg = c & 31, col8 = (seg & 15) * 8;
    if (seg < 16) {
      copy16_g2l(&A0[r * 256 + col8], nf + ((size_t)(b * Nn + m0 + r)) * Uf + col8);
    } else {
      const float* ap = agg + ((size_t)(b * Nn + m0 + r)) * Uf + col8;
      bf16_t* dp = &A0[r * 256 + 128 + col8];
      float4 f0 = *(const float4*)ap, f1 = *(const float4*)(ap + 4);
      dp[0] = (bf16_t)f0.x; dp[1] = (bf16_t)f0.y; dp[2] = (bf16_t)f0.z; dp[3] = (bf16_t)f0.w;
      dp[4] = (bf16_t)f1.x; dp[5] = (bf16_t)f1.y; dp[6] = (bf16_t)f1.z; dp[7] = (bf16_t)f1.w;
    }
  }
  async_wait();
  __syncthreads();
  const int n0 = wave * 32, hi = lane >> 4, lo = lane & 15;
  { // stage 1: h1 = relu(norm1(nc@Wg1 + bg1)) -> A1
    v8f c0 = {}, c1 = {};
    gemm2<256>(A0, Wg1T, lane, n0, c0, c1);
    #pragma unroll
    for (int t = 0; t < 2; ++t) {
      const v8f acc = t ? c1 : c0;
      const int n = n0 + t * 16 + lo;
      const float bn = bg1[n], s = st1[n], tt = st1[Uf + n];
      #pragma unroll
      for (int r8 = 0; r8 < 8; ++r8)
        A1[(r8 + hi * 8) * 128 + n] = (bf16_t)fmaxf(s * (acc[r8] + bn) + tt, 0.f);
    }
  }
  __syncthreads();
  { // stage 2: h3 = norm3(relu(norm2(h1@Wg2 + bg2))) -> A0 (reused, 16x128)
    v8f c0 = {}, c1 = {};
    gemm2<128>(A1, Wg2T, lane, n0, c0, c1);
    #pragma unroll
    for (int t = 0; t < 2; ++t) {
      const v8f acc = t ? c1 : c0;
      const int n = n0 + t * 16 + lo;
      const float bn = bg2[n], s = st2[n], tt = st2[Uf + n];
      const float s3 = st3[n], t3 = st3[Uf + n];
      #pragma unroll
      for (int r8 = 0; r8 < 8; ++r8) {
        float y = fmaxf(s * (acc[r8] + bn) + tt, 0.f);
        A0[(r8 + hi * 8) * 128 + n] = (bf16_t)(s3 * y + t3);
      }
    }
  }
  __syncthreads();
  { // stage 3: nf_out = relu(h3@Wu + bu)
    v8f c0 = {}, c1 = {};
    gemm2<128>(A0, WuT, lane, n0, c0, c1);
    #pragma unroll
    for (int t = 0; t < 2; ++t) {
      const v8f acc = t ? c1 : c0;
      const int n = n0 + t * 16 + lo;
      const float bn = bu[n];
      #pragma unroll
      for (int r8 = 0; r8 < 8; ++r8) {
        const int m = r8 + hi * 8;
        nf_out[((size_t)(b * Nn + m0 + m)) * Uf + n] = (bf16_t)fmaxf(acc[r8] + bn, 0.f);
      }
    }
  }
}

// ---------------- pooling + head ----------------
__global__ void mean_kernel(const bf16_t* __restrict__ nf, float* __restrict__ g) {
  int i = blockIdx.x * 256 + threadIdx.x;
  if (i >= Nb * Uf) return;
  int b = i / Uf, c = i % Uf;
  const bf16_t* p = nf + ((size_t)b * Nn) * Uf + c;
  float s = 0.f;
  for (int n = 0; n < Nn; ++n) s += (float)p[(size_t)n * Uf];
  g[i] = s * (1.f / (float)Nn);
}

__global__ __launch_bounds__(1024) void head_kernel(
    const float* __restrict__ g, const float* __restrict__ Wp1, const float* __restrict__ bp1,
    const float* __restrict__ Wp2, const float* __restrict__ bp2, float* __restrict__ out) {
  __shared__ float g1[Nb * Uf];
  const int t = threadIdx.x;
  if (t < Nb * Uf) {
    int b = t >> 7, u = t & 127;
    float s = bp1[u];
    for (int f = 0; f < Uf; ++f) s += g[b * Uf + f] * Wp1[f * Uf + u];
    g1[t] = fmaxf(s, 0.f);
  }
  __syncthreads();
  if (t < Nb * 64) {
    int b = t >> 6, u = t & 63;
    float s = bp2[u];
    for (int f = 0; f < Uf; ++f) s += g1[b * Uf + f] * Wp2[f * 64 + u];
    s = fmaxf(s, 0.f);
    out[(b * 3 + 0) * 64 + u] = s;
    out[(b * 3 + 1) * 64 + u] = s;
    out[(b * 3 + 2) * 64 + u] = s;
  }
}

extern "C" void kernel_launch(void* const* d_in, const int* in_sizes, int n_in,
                              void* d_out, int out_size, void* d_ws, size_t ws_size,
                              hipStream_t stream) {
  (void)in_sizes; (void)n_in; (void)out_size; (void)ws_size;
  const float* nodes = (const float*)d_in[0];
  const float* edges = (const float*)d_in[1];
  const int*   eidx  = (const int*)d_in[2];
  const int*   erev  = (const int*)d_in[3];
  const float* W_e   = (const float*)d_in[4];
  const float* b_e   = (const float*)d_in[5];
  const float* W_g1  = (const float*)d_in[6];
  const float* b_g1  = (const float*)d_in[7];
  const float* W_g2  = (const float*)d_in[8];
  const float* b_g2  = (const float*)d_in[9];
  const float* W_u   = (const float*)d_in[10];
  const float* b_u   = (const float*)d_in[11];
  const float* nparm = (const float*)d_in[12];
  const float* W_p1  = (const float*)d_in[13];
  const float* b_p1  = (const float*)d_in[14];
  const float* W_p2  = (const float*)d_in[15];
  const float* b_p2  = (const float*)d_in[16];
  float* out = (float*)d_out;

  char* w = (char*)d_ws;
  auto carve = [&](size_t bytes) -> char* {
    char* p = w; w += (bytes + 255) & ~(size_t)255; return p;
  };
  bf16_t* nfbuf0 = (bf16_t*)carve((size_t)Nb * Nn * Uf * 2);
  bf16_t* nfbuf1 = (bf16_t*)carve((size_t)Nb * Nn * Uf * 2);
  bf16_t* efbuf0 = (bf16_t*)carve((size_t)Nb * Ne * Uf * 2);
  bf16_t* efbuf1 = (bf16_t*)carve((size_t)Nb * Ne * Uf * 2);
  float*  agg    = (float*)carve((size_t)Nb * Nn * Uf * 4);
  float*  g      = (float*)carve((size_t)Nb * Uf * 4);
  bf16_t* WeT    = (bf16_t*)carve((size_t)Nd * Uf * 384 * 2);
  bf16_t* Wg1T   = (bf16_t*)carve((size_t)Nd * Uf * 256 * 2);
  bf16_t* Wg2T   = (bf16_t*)carve((size_t)Nd * Uf * 128 * 2);
  bf16_t* WuT    = (bf16_t*)carve((size_t)Nd * Uf * 128 * 2);
  float*  st     = (float*)carve((size_t)Nd * 4 * 2 * Uf * 4);

  for (int l = 0; l < Nd; ++l) {
    pack_wT_kernel<<<(384 * 128 + 255) / 256, 256, 0, stream>>>(
        W_e + (size_t)l * 384 * 128, WeT + (size_t)l * 128 * 384, 384, 128);
    pack_wT_kernel<<<(256 * 128 + 255) / 256, 256, 0, stream>>>(
        W_g1 + (size_t)l * 256 * 128, Wg1T + (size_t)l * 128 * 256, 256, 128);
    pack_wT_kernel<<<(128 * 128 + 255) / 256, 256, 0, stream>>>(
        W_g2 + (size_t)l * 128 * 128, Wg2T + (size_t)l * 128 * 128, 128, 128);
    pack_wT_kernel<<<(128 * 128 + 255) / 256, 256, 0, stream>>>(
        W_u + (size_t)l * 128 * 128, WuT + (size_t)l * 128 * 128, 128, 128);
  }
  norm_prep_kernel<<<(Nd * 4 * Uf + 255) / 256, 256, 0, stream>>>(nparm, st);
  cvt_bf16_kernel<<<(Nb * Nn * Uf + 255) / 256, 256, 0, stream>>>(nodes, nfbuf0, Nb * Nn * Uf);
  cvt_bf16_kernel<<<(Nb * Ne * Uf + 255) / 256, 256, 0, stream>>>(edges, efbuf0, Nb * Ne * Uf);

  bf16_t* nfb[2] = { nfbuf0, nfbuf1 };
  bf16_t* efb[2] = { efbuf0, efbuf1 };
  for (int l = 0; l < Nd; ++l) {
    bf16_t* nf_in  = nfb[l & 1];
    bf16_t* nf_out = nfb[(l + 1) & 1];
    bf16_t* ef_in  = efb[l & 1];
    bf16_t* ef_out = efb[(l + 1) & 1];
    const int n4 = Nb * Nn * Uf / 4;
    zero_kernel<<<(n4 + 255) / 256, 256, 0, stream>>>((float4*)agg, n4);
    edge_mlp_kernel<<<Nb * (Ne / 64), 128, 0, stream>>>(
        nf_in, ef_in, eidx, erev,
        WeT + (size_t)l * 128 * 384, b_e + (size_t)l * Uf,
        st + (size_t)(l * 4 + 0) * 2 * Uf,
        ef_out, agg);
    node_mlp_kernel<<<Nb * (Nn / 16), 128, 0, stream>>>(
        nf_in, agg,
        Wg1T + (size_t)l * 128 * 256, b_g1 + (size_t)l * Uf, st + (size_t)(l * 4 + 1) * 2 * Uf,
        Wg2T + (size_t)l * 128 * 128, b_g2 + (size_t)l * Uf, st + (size_t)(l * 4 + 2) * 2 * Uf,
        st + (size_t)(l * 4 + 3) * 2 * Uf,
        WuT + (size_t)l * 128 * 128, b_u + (size_t)l * Uf,
        nf_out);
  }
  mean_kernel<<<(Nb * Uf + 255) / 256, 256, 0, stream>>>(nfb[Nd & 1], g);
  head_kernel<<<1, 1024, 0, stream>>>(g, W_p1, b_p1, W_p2, b_p2, out);
}
